// HybridMoE_12120397709901
// MI455X (gfx1250) — compile-verified, matrix operand order checked by reading, not run on hardware
//
#include <hip/hip_runtime.h>
#include <hip/hip_bf16.h>
#include <cstddef>
#include <cstdint>

// ---------------- problem constants (match reference) ----------------
constexpr int DIMV = 1024;   // model dim
constexpr int HIDV = 512;    // routed expert hidden
constexpr int SHID = 1024;   // shared expert hidden (HID * N_SHARED)
constexpr int NE   = 16;     // routed experts
constexpr int TT   = 8192;   // tokens (B*S)
constexpr int MT   = 32;     // token tile (two stacked 16x16 WMMA M-halves)
constexpr int SSTRIDE = DIMV + 8;  // 1032 bf16 rows: 2064B, 16B-aligned, bank-spread

typedef __attribute__((ext_vector_type(16))) __bf16 v16bf;
typedef __attribute__((ext_vector_type(8)))  float  v8f;

union BF16x16 { v16bf v; uint4 q[2]; };
union BF16x4  { __bf16 h[4]; uint2 u; };

static __device__ __forceinline__ v8f wmma_bf16(v16bf a, v16bf b, v8f c) {
  return __builtin_amdgcn_wmma_f32_16x16x32_bf16(
      false, a, false, b, (short)0, c, false, false);
}

static __device__ __forceinline__ float silu_f(float v) {
  return v / (1.f + __expf(-v));
}

// A fragment (16x32 bf16) from LDS rows [rowbase, rowbase+16), ISA 7.12.2 layout:
// lanes 0-15: M=lane, k={kb+0..7, kb+16..23}; lanes 16-31: M=lane-16, k={kb+8..15, kb+24..31}
static __device__ __forceinline__ v16bf load_a_lds(const __bf16* S, int rowbase,
                                                   int kbase, int lane) {
  const int m = rowbase + (lane & 15);
  const int g = (lane >> 4) * 8;
  const __bf16* row = S + m * SSTRIDE;
  BF16x16 u;
  u.q[0] = *(const uint4*)(row + kbase + g);        // ds_load_b128
  u.q[1] = *(const uint4*)(row + kbase + 16 + g);   // ds_load_b128
  return u.v;
}

// B fragment (32x16 bf16) from a pre-packed 1KB LDS panel: lane's 32B is contiguous.
static __device__ __forceinline__ v16bf load_b_lds(const __bf16* panel, int lane) {
  const __bf16* p = panel + lane * 16;
  BF16x16 u;
  u.q[0] = *(const uint4*)p;
  u.q[1] = *(const uint4*)(p + 8);
  return u.v;
}

// Async-copy one 1KB packed B panel global->LDS (CDNA5 ASYNCcnt path).
// Each lane moves its own 32B with two b128 async ops (offset applies to both sides).
static __device__ __forceinline__ void async_copy_panel(uint32_t lds_byte,
                                                        const __bf16* gpanel, int lane) {
  uint32_t l = lds_byte + (uint32_t)(lane * 32);
  uint64_t g = (uint64_t)(uintptr_t)gpanel + (uint64_t)(lane * 32);
  asm volatile(
      "global_load_async_to_lds_b128 %0, %1, off\n\t"
      "global_load_async_to_lds_b128 %0, %1, off offset:16"
      :: "v"(l), "v"(g) : "memory");
}
#define WAIT_ASYNC_LE2() asm volatile("s_wait_asynccnt 0x2" ::: "memory")
#define WAIT_ASYNC_0()   asm volatile("s_wait_asynccnt 0x0" ::: "memory")

// ---------------- kernel 0: pack f32 weights -> bf16 WMMA-B fragment order --------
// Packed layout: frag (mat, nt, ks) -> 1KB block of 32 lanes x 16 bf16, lane-contiguous.
__global__ __launch_bounds__(256) void pack_b_kernel(
    const float* __restrict__ W, __bf16* __restrict__ P, int K, int N, int FP) {
  const int tid = threadIdx.x, wave = tid >> 5, lane = tid & 31;
  const int f  = blockIdx.x * 8 + wave;
  const int KS = K / 32;
  const int mat = f / FP, r = f % FP;
  const int nt = r / KS, ks = r % KS;
  const float* Wm = W + (size_t)mat * K * N;
  __bf16*      Pm = P + (size_t)mat * K * N;
  const int n = nt * 16 + (lane & 15);
  const int g = (lane >> 4) * 8;
  const float* p0 = Wm + (size_t)(ks * 32 + g) * N + n;
  const float* p1 = p0 + (size_t)16 * N;
  BF16x16 u;
#pragma unroll
  for (int j = 0; j < 8; ++j) {
    u.v[j]     = (__bf16)p0[(size_t)j * N];
    u.v[j + 8] = (__bf16)p1[(size_t)j * N];
  }
  uint4* dst = (uint4*)(Pm + ((size_t)r * 32 + lane) * 16);  // 32B/lane, coalesced
  dst[0] = u.q[0];
  dst[1] = u.q[1];
}

// ---------------- kernel 1: gating (top-2 softmax renorm + scatter lists) ---------
__global__ __launch_bounds__(256) void moe_gate_kernel(
    const float* __restrict__ x, const float* __restrict__ gw,
    int* __restrict__ counts, int* __restrict__ tok_idx, float* __restrict__ tok_w) {
  __shared__ float lg[8][NE];
  const int tid = threadIdx.x, wave = tid >> 5, lane = tid & 31;
  const int t = blockIdx.x * 8 + wave;           // one wave per token
  const int e = lane & 15, half = lane >> 4;
  const float* xr = x + (size_t)t * DIMV + half * (DIMV / 2);
  const float* gc = gw + e + (size_t)half * (DIMV / 2) * NE;
  float acc = 0.f;
  for (int d = 0; d < DIMV / 2; ++d) acc += xr[d] * gc[(size_t)d * NE];
  acc += __shfl_down(acc, 16);                   // fold halves (wave32)
  if (lane < 16) lg[wave][e] = acc;
  __syncthreads();
  if (lane == 0) {
    float l0 = -1e30f, l1 = -1e30f; int i0 = 0, i1 = 0;
    for (int i = 0; i < NE; ++i) {
      float v = lg[wave][i];
      if (v > l0)      { l1 = l0; i1 = i0; l0 = v; i0 = i; }
      else if (v > l1) { l1 = v; i1 = i; }
    }
    float w0 = 1.f / (1.f + __expf(l1 - l0));    // top-2 softmax renorm
    float w1v = 1.f - w0;
    int p0 = atomicAdd(&counts[i0], 1);
    tok_idx[(size_t)i0 * TT + p0] = t; tok_w[(size_t)i0 * TT + p0] = w0;
    int p1 = atomicAdd(&counts[i1], 1);
    tok_idx[(size_t)i1 * TT + p1] = t; tok_w[(size_t)i1 * TT + p1] = w1v;
  }
}

// ---------------- kernel 2: shared expert (M=32), writes all of d_out -------------
__global__ __launch_bounds__(256) void moe_shared_kernel(
    const float* __restrict__ x, const __bf16* __restrict__ Pw1,
    const __bf16* __restrict__ Pw2, float* __restrict__ out) {
  __shared__ __bf16 S[MT * SSTRIDE];       // 66KB: X tile, then reused as H tile
  __shared__ __bf16 Bbuf[8 * 2 * 512];     // 16KB: per-wave double-buffered B panels
  const int tid = threadIdx.x, wave = tid >> 5, lane = tid & 31;
  const int t0 = blockIdx.x * MT;
  const int n_lo = lane & 15, mbase = (lane >> 4) * 8;
  const uint32_t bbase = (uint32_t)(uintptr_t)(&Bbuf[0]) + (uint32_t)(wave * 2048);
  const __bf16* bptr = &Bbuf[wave * 1024];

  // stage X tile, f32 -> bf16, float4 reads / 8B LDS stores
  for (int i4 = tid; i4 < MT * (DIMV / 4); i4 += 256) {
    int r = i4 >> 8, c4 = (i4 & 255) * 4;
    float4 v = *(const float4*)(x + (size_t)(t0 + r) * DIMV + c4);
    BF16x4 h; h.h[0]=(__bf16)v.x; h.h[1]=(__bf16)v.y; h.h[2]=(__bf16)v.z; h.h[3]=(__bf16)v.w;
    *(uint2*)&S[r * SSTRIDE + c4] = h.u;
  }
  __syncthreads();

  constexpr int KS = DIMV / 32;            // 32 k-steps
  __bf16 hreg[8][2][8];                    // H kept in regs so S can be reused
#pragma unroll
  for (int j = 0; j < 8; ++j) {            // 64 n-tiles / 8 waves
    const int nt = wave * 8 + j;
    const __bf16* panel = Pw1 + (size_t)nt * KS * 512;
    async_copy_panel(bbase, panel, lane);
    v8f acc0 = {}, acc1 = {};
    int slot = 0;
    for (int ks = 0; ks < KS; ++ks) {
      if (ks + 1 < KS) {
        async_copy_panel(bbase + (uint32_t)((slot ^ 1) * 1024),
                         panel + (size_t)(ks + 1) * 512, lane);
        WAIT_ASYNC_LE2();                  // current panel landed in LDS
      } else {
        WAIT_ASYNC_0();
      }
      v16bf b  = load_b_lds(bptr + slot * 512, lane);
      v16bf a0 = load_a_lds(S, 0,  ks * 32, lane);
      v16bf a1 = load_a_lds(S, 16, ks * 32, lane);
      acc0 = wmma_bf16(a0, b, acc0);
      acc1 = wmma_bf16(a1, b, acc1);       // B reused across both M-halves
      slot ^= 1;
    }
#pragma unroll
    for (int r = 0; r < 8; ++r) {
      hreg[j][0][r] = (__bf16)silu_f(acc0[r]);
      hreg[j][1][r] = (__bf16)silu_f(acc1[r]);
    }
  }
  __syncthreads();                         // everyone done reading X
#pragma unroll
  for (int j = 0; j < 8; ++j) {
    const int nt = wave * 8 + j;
#pragma unroll
    for (int r = 0; r < 8; ++r) {
      S[(mbase + r)      * SSTRIDE + nt * 16 + n_lo] = hreg[j][0][r];
      S[(16 + mbase + r) * SSTRIDE + nt * 16 + n_lo] = hreg[j][1][r];
    }
  }
  __syncthreads();

#pragma unroll 1
  for (int j = 0; j < 8; ++j) {            // OUT = H @ W2, full write of d_out
    const int nt = wave * 8 + j;
    const __bf16* panel = Pw2 + (size_t)nt * KS * 512;   // K = SHID = 1024
    async_copy_panel(bbase, panel, lane);
    v8f acc0 = {}, acc1 = {};
    int slot = 0;
    for (int ks = 0; ks < KS; ++ks) {
      if (ks + 1 < KS) {
        async_copy_panel(bbase + (uint32_t)((slot ^ 1) * 1024),
                         panel + (size_t)(ks + 1) * 512, lane);
        WAIT_ASYNC_LE2();
      } else {
        WAIT_ASYNC_0();
      }
      v16bf b  = load_b_lds(bptr + slot * 512, lane);
      v16bf a0 = load_a_lds(S, 0,  ks * 32, lane);
      v16bf a1 = load_a_lds(S, 16, ks * 32, lane);
      acc0 = wmma_bf16(a0, b, acc0);
      acc1 = wmma_bf16(a1, b, acc1);
      slot ^= 1;
    }
#pragma unroll
    for (int r = 0; r < 8; ++r) {
      out[(size_t)(t0 + mbase + r)      * DIMV + nt * 16 + n_lo] = acc0[r];
      out[(size_t)(t0 + 16 + mbase + r) * DIMV + nt * 16 + n_lo] = acc1[r];
    }
  }
}

// ---------------- kernel 3: routed experts (M=32), gathered tiles, atomic scatter -
__global__ __launch_bounds__(256) void moe_routed_kernel(
    const float* __restrict__ x, const __bf16* __restrict__ Pw1all,
    const __bf16* __restrict__ Pw2all, const int* __restrict__ counts,
    const int* __restrict__ tok_idx, const float* __restrict__ tok_w,
    float* __restrict__ out) {
  const int e = blockIdx.y;
  const int tile = blockIdx.x;
  const int cnt = counts[e];
  if (tile * MT >= cnt) return;            // most blocks exit immediately
  const int rows = min(MT, cnt - tile * MT);

  __shared__ __bf16 S[MT * SSTRIDE];       // 66KB: X tile, then H tile (cols 0..511)
  __shared__ __bf16 Bbuf[8 * 2 * 512];     // 16KB: per-wave async B panels
  __shared__ int   tok[MT];
  __shared__ float wt[MT];

  const int tid = threadIdx.x, wave = tid >> 5, lane = tid & 31;
  const int n_lo = lane & 15, mbase = (lane >> 4) * 8;
  const uint32_t bbase = (uint32_t)(uintptr_t)(&Bbuf[0]) + (uint32_t)(wave * 2048);
  const __bf16* bptr = &Bbuf[wave * 1024];

  if (tid < MT) {
    if (tid < rows) {
      tok[tid] = tok_idx[(size_t)e * TT + tile * MT + tid];
      wt[tid]  = tok_w [(size_t)e * TT + tile * MT + tid];
    } else { tok[tid] = 0; wt[tid] = 0.f; }
  }
  __syncthreads();
  for (int i4 = tid; i4 < MT * (DIMV / 4); i4 += 256) {
    int r = i4 >> 8, c4 = (i4 & 255) * 4;
    BF16x4 h;
    if (r < rows) {
      float4 v = *(const float4*)(x + (size_t)tok[r] * DIMV + c4);
      h.h[0]=(__bf16)v.x; h.h[1]=(__bf16)v.y; h.h[2]=(__bf16)v.z; h.h[3]=(__bf16)v.w;
    } else {
      h.h[0]=(__bf16)0.f; h.h[1]=(__bf16)0.f; h.h[2]=(__bf16)0.f; h.h[3]=(__bf16)0.f;
    }
    *(uint2*)&S[r * SSTRIDE + c4] = h.u;
  }
  __syncthreads();

  const __bf16* Pw1 = Pw1all + (size_t)e * DIMV * HIDV;
  const __bf16* Pw2 = Pw2all + (size_t)e * HIDV * DIMV;

  // Phase 1: H = silu(X @ w1[e]); 32 n-tiles / 8 waves; K = 1024
  constexpr int KS1 = DIMV / 32;           // 32
  __bf16 hreg[4][2][8];
#pragma unroll
  for (int j = 0; j < 4; ++j) {
    const int nt = wave * 4 + j;
    const __bf16* panel = Pw1 + (size_t)nt * KS1 * 512;
    async_copy_panel(bbase, panel, lane);
    v8f acc0 = {}, acc1 = {};
    int slot = 0;
    for (int ks = 0; ks < KS1; ++ks) {
      if (ks + 1 < KS1) {
        async_copy_panel(bbase + (uint32_t)((slot ^ 1) * 1024),
                         panel + (size_t)(ks + 1) * 512, lane);
        WAIT_ASYNC_LE2();
      } else {
        WAIT_ASYNC_0();
      }
      v16bf b  = load_b_lds(bptr + slot * 512, lane);
      v16bf a0 = load_a_lds(S, 0,  ks * 32, lane);
      v16bf a1 = load_a_lds(S, 16, ks * 32, lane);
      acc0 = wmma_bf16(a0, b, acc0);
      acc1 = wmma_bf16(a1, b, acc1);
      slot ^= 1;
    }
#pragma unroll
    for (int r = 0; r < 8; ++r) {
      hreg[j][0][r] = (__bf16)silu_f(acc0[r]);
      hreg[j][1][r] = (__bf16)silu_f(acc1[r]);
    }
  }
  __syncthreads();                         // everyone done reading X
#pragma unroll
  for (int j = 0; j < 4; ++j) {
    const int nt = wave * 4 + j;
#pragma unroll
    for (int r = 0; r < 8; ++r) {
      S[(mbase + r)      * SSTRIDE + nt * 16 + n_lo] = hreg[j][0][r];
      S[(16 + mbase + r) * SSTRIDE + nt * 16 + n_lo] = hreg[j][1][r];
    }
  }
  __syncthreads();

  // Phase 2: out[tok] += wt * (H @ w2[e]); 64 n-tiles / 8 waves; K = 512
  constexpr int KS2 = HIDV / 32;           // 16
#pragma unroll 1
  for (int j = 0; j < 8; ++j) {
    const int nt = wave * 8 + j;
    const __bf16* panel = Pw2 + (size_t)nt * KS2 * 512;
    async_copy_panel(bbase, panel, lane);
    v8f acc0 = {}, acc1 = {};
    int slot = 0;
    for (int ks = 0; ks < KS2; ++ks) {
      if (ks + 1 < KS2) {
        async_copy_panel(bbase + (uint32_t)((slot ^ 1) * 1024),
                         panel + (size_t)(ks + 1) * 512, lane);
        WAIT_ASYNC_LE2();
      } else {
        WAIT_ASYNC_0();
      }
      v16bf b  = load_b_lds(bptr + slot * 512, lane);
      v16bf a0 = load_a_lds(S, 0,  ks * 32, lane);
      v16bf a1 = load_a_lds(S, 16, ks * 32, lane);
      acc0 = wmma_bf16(a0, b, acc0);
      acc1 = wmma_bf16(a1, b, acc1);
      slot ^= 1;
    }
#pragma unroll
    for (int r = 0; r < 8; ++r) {
      int m0 = mbase + r, m1 = 16 + mbase + r;
      if (m0 < rows)
        atomicAdd(&out[(size_t)tok[m0] * DIMV + nt * 16 + n_lo], acc0[r] * wt[m0]);
      if (m1 < rows)
        atomicAdd(&out[(size_t)tok[m1] * DIMV + nt * 16 + n_lo], acc1[r] * wt[m1]);
    }
  }
}

// ---------------- launcher ----------------
extern "C" void kernel_launch(void* const* d_in, const int* in_sizes, int n_in,
                              void* d_out, int out_size, void* d_ws, size_t ws_size,
                              hipStream_t stream) {
  const float* x      = (const float*)d_in[0];
  const float* gate_w = (const float*)d_in[1];
  const float* w1     = (const float*)d_in[2];
  const float* w2     = (const float*)d_in[3];
  const float* sw1    = (const float*)d_in[4];
  const float* sw2    = (const float*)d_in[5];
  float* out = (float*)d_out;

  // workspace: counts | tok_idx | tok_w | packed bf16 weights (~38.8 MB total)
  char* ws = (char*)d_ws;
  int*   counts  = (int*)ws;                    size_t off = 256;
  int*   tok_idx = (int*)(ws + off);            off += (size_t)NE * TT * 4;
  float* tok_w   = (float*)(ws + off);          off += (size_t)NE * TT * 4;
  __bf16* Pw1    = (__bf16*)(ws + off);         off += (size_t)NE * DIMV * HIDV * 2;
  __bf16* Pw2    = (__bf16*)(ws + off);         off += (size_t)NE * HIDV * DIMV * 2;
  __bf16* Psw1   = (__bf16*)(ws + off);         off += (size_t)DIMV * SHID * 2;
  __bf16* Psw2   = (__bf16*)(ws + off);

  hipMemsetAsync(counts, 0, 256, stream);

  // pack f32 weights into bf16 WMMA-B fragment order (once per launch)
  pack_b_kernel<<<2048, 256, 0, stream>>>(w1,  Pw1,  DIMV, HIDV, (HIDV/16)*(DIMV/32));
  pack_b_kernel<<<2048, 256, 0, stream>>>(w2,  Pw2,  HIDV, DIMV, (DIMV/16)*(HIDV/32));
  pack_b_kernel<<< 256, 256, 0, stream>>>(sw1, Psw1, DIMV, SHID, (SHID/16)*(DIMV/32));
  pack_b_kernel<<< 256, 256, 0, stream>>>(sw2, Psw2, SHID, DIMV, (DIMV/16)*(SHID/32));

  moe_gate_kernel<<<TT / 8, 256, 0, stream>>>(x, gate_w, counts, tok_idx, tok_w);
  moe_shared_kernel<<<TT / MT, 256, 0, stream>>>(x, Psw1, Psw2, out);
  moe_routed_kernel<<<dim3(TT / MT, NE), 256, 0, stream>>>(
      x, Pw1, Pw2, counts, tok_idx, tok_w, out);
}
